// EncoderLayer_ffn_moe_6330781794933
// MI455X (gfx1250) — matrix operand order repaired and checked by reading, not compiled, not run on hardware
//
#include <hip/hip_runtime.h>
#include <hip/hip_bf16.h>
#include <math.h>

// ---------------------------------------------------------------------------
// EncoderLayer (pre-LN MHA + top-1 Switch MoE) for gfx1250 (MI455X).
// - All GEMMs on v_wmma_f32_16x16x32_bf16 (bf16 operands, fp32 accum).
// - 128x128 register-blocked GEMM: 8 waves, each wave 32x64 of C
//   (8 WMMA per K-step per wave from 12 LDS b128 fragment loads).
// - A tiles staged global->LDS with gfx1250 async-tensor path
//   (global_load_async_to_lds_b128 + s_wait_asynccnt), B tiles transposed
//   into LDS with packed ds_store_b64.
// - Fused flash attention (scores never hit HBM).
// - MoE gather/scatter routing (top-1 => each token written exactly once).
// ---------------------------------------------------------------------------

typedef __bf16 bf16_t;
typedef __attribute__((ext_vector_type(16))) __bf16 v16bf;
typedef __attribute__((ext_vector_type(8)))  float  v8f;

#define NTOK 4096   // B*T
#define DMODEL 1024
#define DFFN 4096
#define NEXP 8
#define SEQT 1024
#define NHEAD 16

static __device__ __forceinline__ v8f wmma_bf16(v16bf a, v16bf b, v8f c) {
  // D = A(16x32 bf16) * B(32x16 bf16) + C(16x16 f32)
  return __builtin_amdgcn_wmma_f32_16x16x32_bf16(false, a, false, b,
                                                 (short)0, c, false, false);
}

union Frag16 { v16bf v; uint4 u[2]; };

// gfx1250 async memory->LDS copy (ASYNCcnt path), ISA §15.18.3 op 98.
// vdst = per-lane LDS byte address, vaddr = per-lane 64-bit global address.
static __device__ __forceinline__ void async_copy_b128(unsigned int lds_off,
                                                       const void* gptr) {
  asm volatile("global_load_async_to_lds_b128 %0, %1, off"
               :: "v"(lds_off), "v"((unsigned long long)(uintptr_t)gptr)
               : "memory");
}
static __device__ __forceinline__ void wait_async0() {
  asm volatile("s_wait_asynccnt 0x0" ::: "memory");
}

// ---------------------------------------------------------------------------
// fp32 -> bf16 converter (weights)
// ---------------------------------------------------------------------------
__global__ __launch_bounds__(256)
void k_cvt(const float* __restrict__ s, bf16_t* __restrict__ d, size_t n) {
  size_t i = (size_t)blockIdx.x * 256 + threadIdx.x;
  size_t stride = (size_t)gridDim.x * 256;
  for (; i < n; i += stride) d[i] = (bf16_t)s[i];
}

__global__ void k_zero_i32(int* p, int n) {
  int i = blockIdx.x * blockDim.x + threadIdx.x;
  if (i < n) p[i] = 0;
}

// ---------------------------------------------------------------------------
// LayerNorm (torch-style: unbiased std, divide by (std+eps)) -> bf16
// ---------------------------------------------------------------------------
__global__ __launch_bounds__(256)
void k_ln(const float* __restrict__ x, const float* __restrict__ ga,
          const float* __restrict__ gbeta, bf16_t* __restrict__ outH) {
  __shared__ float red[256];
  __shared__ float s_mean, s_inv;
  const int n = blockIdx.x, tid = threadIdx.x;
  const float* row = x + (size_t)n * DMODEL;
  float s = 0.f;
  for (int c = tid; c < DMODEL; c += 256) s += row[c];
  red[tid] = s; __syncthreads();
  for (int st = 128; st > 0; st >>= 1) {
    if (tid < st) red[tid] += red[tid + st];
    __syncthreads();
  }
  if (tid == 0) s_mean = red[0] * (1.f / DMODEL);
  __syncthreads();
  float mean = s_mean;
  float s2 = 0.f;
  for (int c = tid; c < DMODEL; c += 256) { float d = row[c] - mean; s2 += d * d; }
  red[tid] = s2; __syncthreads();
  for (int st = 128; st > 0; st >>= 1) {
    if (tid < st) red[tid] += red[tid + st];
    __syncthreads();
  }
  if (tid == 0) s_inv = 1.f / (sqrtf(red[0] * (1.f / (DMODEL - 1))) + 1e-6f);
  __syncthreads();
  float inv = s_inv;
  for (int c = tid; c < DMODEL; c += 256)
    outH[(size_t)n * DMODEL + c] = (bf16_t)(ga[c] * (row[c] - mean) * inv + gbeta[c]);
}

// ---------------------------------------------------------------------------
// 128x128-tile WMMA GEMM. 256 threads = 8 waves; wave (wm,wn) owns 32x64 of C.
// Per K-step: threads 0-127 stage transposed B (packed ds_store_b64),
// threads 128-255 issue async global->LDS copies of the A tile.
//   MODE 0: C = A@B + bias          (+optional residual, f32 and/or bf16 out)
//   MODE 1: expert GEMM1 (gather x1h rows) -> relu -> h (bf16)
//   MODE 2: expert GEMM2 (rows of h) -> d_out[tok] = x1f[tok] + p*(acc+bias)
// ---------------------------------------------------------------------------
template <int MODE>
__global__ __launch_bounds__(256)
void k_gemm(const bf16_t* __restrict__ A, int lda,
            const bf16_t* __restrict__ Bw, int ldb,
            const float* __restrict__ bias,
            int M, int Ndim, int Kdim,
            const float* __restrict__ resF,
            float* __restrict__ outF,
            bf16_t* __restrict__ outH,
            const int* __restrict__ counts,
            const int* __restrict__ offs,
            const int* __restrict__ lists,
            const float* __restrict__ top1p,
            bf16_t* __restrict__ hbuf,
            const bf16_t* __restrict__ x1h,
            const float* __restrict__ x1f,
            float* __restrict__ outx2,
            int Ntot) {
  __shared__ alignas(16) bf16_t As[128][40];  // [row][k], stride 80B
  __shared__ alignas(16) bf16_t Bt[128][40];  // [col][k], stride 80B
  __shared__ int   toks_s[128];
  __shared__ float pv_s[128];

  const int tid  = threadIdx.x;
  const int m0   = blockIdx.x * 128;
  const int nblk = blockIdx.y * 128;
  const int e    = blockIdx.z;

  int cntE = 0, offE = 0;
  if (MODE != 0) {
    cntE = counts[e];
    if (m0 >= cntE) return;      // uniform per block
    offE = offs[e];
    if (tid < 128) {
      int idx = m0 + tid; if (idx > cntE - 1) idx = cntE - 1;
      int tok = lists[e * Ntot + idx];
      toks_s[tid] = tok;
      if (MODE == 2) pv_s[tid] = top1p[tok];
    }
    __syncthreads();
  }

  const bf16_t* Bmat  = Bw + (size_t)e * Kdim * Ndim;
  const float*  biasp = bias + (size_t)e * Ndim;

  const int w  = tid >> 5;
  const int wm = w & 3;        // 4 row strips of 32
  const int wn = w >> 2;       // 2 col strips of 64
  const int l  = tid & 31;
  const int lh = l & 15;
  const bool hi = l >= 16;

  v8f acc[2][4];
  const v8f vz = {0.f, 0.f, 0.f, 0.f, 0.f, 0.f, 0.f, 0.f};
#pragma unroll
  for (int ah = 0; ah < 2; ++ah)
#pragma unroll
    for (int nt = 0; nt < 4; ++nt) acc[ah][nt] = vz;

  for (int k0 = 0; k0 < Kdim; k0 += 32) {
    if (tid < 128) {
      // ---- stage B tile transposed: 32 k-rows x 128 cols -> Bt[col][k]
      // each thread: 4x8 sub-tile (rows 4*kq..+3, cols n0..n0+7),
      // re-packed into 8 ds_store_b64.
      int kq = tid >> 4;           // 0..7 -> k rows 4*kq..4*kq+3
      int n0 = (tid & 15) << 3;    // 0..120
      const bf16_t* bb0 = Bmat + (size_t)(k0 + 4 * kq) * ldb + nblk + n0;
      uint4 r0 = *(const uint4*)(bb0);
      uint4 r1 = *(const uint4*)(bb0 + ldb);
      uint4 r2 = *(const uint4*)(bb0 + 2 * ldb);
      uint4 r3 = *(const uint4*)(bb0 + 3 * ldb);
      const unsigned short* p0 = (const unsigned short*)&r0;
      const unsigned short* p1 = (const unsigned short*)&r1;
      const unsigned short* p2 = (const unsigned short*)&r2;
      const unsigned short* p3 = (const unsigned short*)&r3;
#pragma unroll
      for (int i = 0; i < 8; ++i) {
        uint2 pk;
        pk.x = (unsigned)p0[i] | ((unsigned)p1[i] << 16);
        pk.y = (unsigned)p2[i] | ((unsigned)p3[i] << 16);
        *(uint2*)&Bt[n0 + i][4 * kq] = pk;
      }
    } else {
      // ---- stage A tile (128 rows x 32 halfs) via async global->LDS
      int t2 = tid - 128;
#pragma unroll
      for (int i = 0; i < 4; ++i) {
        int idx = t2 * 4 + i;          // 0..511
        int r   = idx >> 2;            // 0..127
        int c0  = (idx & 3) << 3;      // 0,8,16,24
        const bf16_t* srcRow;
        if (MODE == 0) {
          int gr = m0 + r; if (gr >= M) gr = M - 1;
          srcRow = A + (size_t)gr * lda;
        } else if (MODE == 1) {
          srcRow = x1h + (size_t)toks_s[r] * lda;
        } else {
          srcRow = hbuf + (size_t)(offE + m0 + r) * lda;  // h padded 128 rows
        }
        async_copy_b128((unsigned int)(uintptr_t)&As[r][c0], srcRow + k0 + c0);
      }
    }
    wait_async0();
    __syncthreads();

    // ---- fragments + 8 WMMA per wave
    // A frag: lanes 0-15 row m, K=0..7 & 16..23; lanes 16-31: K=8..15 & 24..31
    Frag16 a0, a1;
    int am = wm * 32 + lh;
    int ab = hi ? 8 : 0;
    a0.u[0] = *(const uint4*)&As[am][ab];
    a0.u[1] = *(const uint4*)&As[am][ab + 16];
    a1.u[0] = *(const uint4*)&As[am + 16][ab];
    a1.u[1] = *(const uint4*)&As[am + 16][ab + 16];
#pragma unroll
    for (int nt = 0; nt < 4; ++nt) {
      Frag16 fb;
      int bn = wn * 64 + nt * 16 + lh;
      int bb = hi ? 8 : 0;
      fb.u[0] = *(const uint4*)&Bt[bn][bb];
      fb.u[1] = *(const uint4*)&Bt[bn][bb + 16];
      acc[0][nt] = wmma_bf16(a0.v, fb.v, acc[0][nt]);
      acc[1][nt] = wmma_bf16(a1.v, fb.v, acc[1][nt]);
    }
    __syncthreads();
  }

  // ---- epilogue (C layout: lanes 0-15 -> M=j, lanes 16-31 -> M=j+8; N=lh)
#pragma unroll
  for (int ah = 0; ah < 2; ++ah) {
#pragma unroll
    for (int nt = 0; nt < 4; ++nt) {
      int c = nblk + wn * 64 + nt * 16 + lh;
      float bz = biasp[c];
#pragma unroll
      for (int j = 0; j < 8; ++j) {
        int rloc = wm * 32 + ah * 16 + j + (hi ? 8 : 0);
        float v = acc[ah][nt][j] + bz;
        if (MODE == 0) {
          int gr = m0 + rloc;
          if (gr < M) {
            if (resF) v += resF[(size_t)gr * Ndim + c];
            if (outF) outF[(size_t)gr * Ndim + c] = v;
            if (outH) outH[(size_t)gr * Ndim + c] = (bf16_t)v;
          }
        } else if (MODE == 1) {
          if (m0 + rloc < cntE) {
            v = fmaxf(v, 0.0f);
            hbuf[(size_t)(offE + m0 + rloc) * Ndim + c] = (bf16_t)v;
          }
        } else {
          if (m0 + rloc < cntE) {
            int tok = toks_s[rloc];
            outx2[(size_t)tok * Ndim + c] =
                x1f[(size_t)tok * Ndim + c] + pv_s[rloc] * v;
          }
        }
      }
    }
  }
}

// ---------------------------------------------------------------------------
// Fused flash attention. Grid (T/64, B*H), 128 threads = 4 waves, each wave
// owns 16 query rows. Online softmax; P re-laid out via LDS for P@V WMMA.
// ---------------------------------------------------------------------------
__global__ __launch_bounds__(128)
void k_attn(const bf16_t* __restrict__ Qh, const bf16_t* __restrict__ Kh,
            const bf16_t* __restrict__ Vh, const unsigned char* __restrict__ mask,
            bf16_t* __restrict__ attnH) {
  __shared__ alignas(16) bf16_t Qs[64][80];
  __shared__ alignas(16) bf16_t Ks[64][80];   // [key][dh]
  __shared__ alignas(16) bf16_t Vt[64][80];   // [dh][key]
  __shared__ alignas(16) bf16_t Ps[64][80];   // [qrow][key]

  const int tid = threadIdx.x;
  const int w = tid >> 5, l = tid & 31, lh = l & 15;
  const bool hi = l >= 16;
  const int bh = blockIdx.y;
  const int b = bh >> 4, hh = bh & (NHEAD - 1);
  const int q0 = blockIdx.x * 64;

  // stage Q block (64 x 64 bf16)
#pragma unroll
  for (int it = 0; it < 4; ++it) {
    int idx = (tid * 4 + it) * 8;
    int r = idx >> 6, c = idx & 63;
    *(uint4*)&Qs[r][c] =
        *(const uint4*)(Qh + ((size_t)(b * SEQT + q0 + r)) * DMODEL + hh * 64 + c);
  }
  __syncthreads();

  Frag16 qf[2];
  const int qm = w * 16 + lh;
  const int ab = hi ? 8 : 0;
#pragma unroll
  for (int ks = 0; ks < 2; ++ks) {
    qf[ks].u[0] = *(const uint4*)&Qs[qm][ks * 32 + ab];
    qf[ks].u[1] = *(const uint4*)&Qs[qm][ks * 32 + ab + 16];
  }

  float mrun[8], lrun[8];
  v8f oacc[4];
  const v8f vz = {0.f, 0.f, 0.f, 0.f, 0.f, 0.f, 0.f, 0.f};
#pragma unroll
  for (int j = 0; j < 8; ++j) { mrun[j] = -3.0e38f; lrun[j] = 0.f; }
#pragma unroll
  for (int nt = 0; nt < 4; ++nt) oacc[nt] = vz;

  for (int kc = 0; kc < SEQT / 64; ++kc) {
    __syncthreads();  // protect Ks/Vt/Ps from previous iteration readers
    // stage K chunk [key][dh] and V chunk transposed [dh][key]
#pragma unroll
    for (int it = 0; it < 4; ++it) {
      int idx = (tid * 4 + it) * 8;
      int r = idx >> 6, c = idx & 63;
      size_t grow = ((size_t)(b * SEQT + kc * 64 + r)) * DMODEL + hh * 64 + c;
      *(uint4*)&Ks[r][c] = *(const uint4*)(Kh + grow);
      uint4 vv = *(const uint4*)(Vh + grow);
      const bf16_t* pv = (const bf16_t*)&vv;
#pragma unroll
      for (int i = 0; i < 8; ++i) Vt[c + i][r] = pv[i];
    }
    __syncthreads();

    // S = Q @ K^T  (16 rows x 64 keys per wave)
    v8f sacc[4];
#pragma unroll
    for (int nt = 0; nt < 4; ++nt) sacc[nt] = vz;
#pragma unroll
    for (int ks = 0; ks < 2; ++ks) {
#pragma unroll
      for (int nt = 0; nt < 4; ++nt) {
        Frag16 kf;
        int kn = nt * 16 + lh;
        int db = ks * 32 + (hi ? 8 : 0);
        kf.u[0] = *(const uint4*)&Ks[kn][db];
        kf.u[1] = *(const uint4*)&Ks[kn][db + 16];
        sacc[nt] = wmma_bf16(qf[ks].v, kf.v, sacc[nt]);
      }
    }

    // scale + mask (replace with -1e9 where masked, per reference)
    float pbuf[4][8];
#pragma unroll
    for (int nt = 0; nt < 4; ++nt) {
      int key = kc * 64 + nt * 16 + lh;
      bool mok = mask[b * SEQT + key] != 0;
#pragma unroll
      for (int j = 0; j < 8; ++j) {
        float s = sacc[nt][j] * 0.125f;    // 1/sqrt(64)
        pbuf[nt][j] = mok ? s : -1e9f;
      }
    }

    // online softmax per row (rows replicated across 16-lane halves)
#pragma unroll
    for (int j = 0; j < 8; ++j) {
      float mx = -3.0e38f;
#pragma unroll
      for (int nt = 0; nt < 4; ++nt) mx = fmaxf(mx, pbuf[nt][j]);
#pragma unroll
      for (int off = 1; off < 16; off <<= 1)
        mx = fmaxf(mx, __shfl_xor(mx, off, 16));
      float mn = fmaxf(mrun[j], mx);
      float alpha = __expf(mrun[j] - mn);
      float rs = 0.f;
#pragma unroll
      for (int nt = 0; nt < 4; ++nt) {
        float p = __expf(pbuf[nt][j] - mn);
        pbuf[nt][j] = p;
        rs += p;
      }
#pragma unroll
      for (int off = 1; off < 16; off <<= 1) rs += __shfl_xor(rs, off, 16);
      lrun[j] = lrun[j] * alpha + rs;
      mrun[j] = mn;
#pragma unroll
      for (int nt = 0; nt < 4; ++nt) oacc[nt][j] *= alpha;
    }

    // write P (C-layout regs -> row-major LDS) for A-fragment reload
    {
      int pr = w * 16 + (hi ? 8 : 0);
#pragma unroll
      for (int nt = 0; nt < 4; ++nt)
#pragma unroll
        for (int j = 0; j < 8; ++j)
          Ps[pr + j][nt * 16 + lh] = (bf16_t)pbuf[nt][j];
    }
    __syncthreads();

    // O += P @ V
#pragma unroll
    for (int ks = 0; ks < 2; ++ks) {
      Frag16 pf;
      int pm = w * 16 + lh;
      int pb = ks * 32 + (hi ? 8 : 0);
      pf.u[0] = *(const uint4*)&Ps[pm][pb];
      pf.u[1] = *(const uint4*)&Ps[pm][pb + 16];
#pragma unroll
      for (int nt = 0; nt < 4; ++nt) {
        Frag16 vf;
        int vn = nt * 16 + lh;
        int vb = ks * 32 + (hi ? 8 : 0);
        vf.u[0] = *(const uint4*)&Vt[vn][vb];
        vf.u[1] = *(const uint4*)&Vt[vn][vb + 16];
        oacc[nt] = wmma_bf16(pf.v, vf.v, oacc[nt]);
      }
    }
  }

  // epilogue: O / l, written to [token][h*64+dh] so the Wo GEMM reads it flat
#pragma unroll
  for (int nt = 0; nt < 4; ++nt) {
    int col = hh * 64 + nt * 16 + lh;
#pragma unroll
    for (int j = 0; j < 8; ++j) {
      int row = q0 + w * 16 + j + (hi ? 8 : 0);
      float v = oacc[nt][j] / lrun[j];
      attnH[((size_t)(b * SEQT + row)) * DMODEL + col] = (bf16_t)v;
    }
  }
}

// ---------------------------------------------------------------------------
// Gate: softmax(x1 @ gW + gb), top-1 routing, per-expert lists + counts
// ---------------------------------------------------------------------------
__global__ __launch_bounds__(256)
void k_gate(const float* __restrict__ x1f, const float* __restrict__ gW,
            const float* __restrict__ gb, int* __restrict__ top1_idx,
            float* __restrict__ top1_p, int* __restrict__ counts,
            int* __restrict__ lists) {
  __shared__ float red[256];
  __shared__ float logit[NEXP];
  const int n = blockIdx.x, tid = threadIdx.x;
  const float* row = x1f + (size_t)n * DMODEL;
  float acc[NEXP];
#pragma unroll
  for (int e = 0; e < NEXP; ++e) acc[e] = 0.f;
  for (int dd = tid; dd < DMODEL; dd += 256) {
    float xv = row[dd];
    const float* gr = gW + (size_t)dd * NEXP;
#pragma unroll
    for (int e = 0; e < NEXP; ++e) acc[e] += xv * gr[e];
  }
  for (int e = 0; e < NEXP; ++e) {
    red[tid] = acc[e]; __syncthreads();
    for (int st = 128; st > 0; st >>= 1) {
      if (tid < st) red[tid] += red[tid + st];
      __syncthreads();
    }
    if (tid == 0) logit[e] = red[0] + gb[e];
    __syncthreads();
  }
  if (tid == 0) {
    float mx = logit[0];
#pragma unroll
    for (int e = 1; e < NEXP; ++e) mx = fmaxf(mx, logit[e]);
    float pr[NEXP], sum = 0.f;
#pragma unroll
    for (int e = 0; e < NEXP; ++e) { pr[e] = __expf(logit[e] - mx); sum += pr[e]; }
    int bi = 0; float bp = pr[0];
#pragma unroll
    for (int e = 1; e < NEXP; ++e)
      if (pr[e] > bp) { bp = pr[e]; bi = e; }   // first max wins, like argmax
    bp /= sum;
    top1_idx[n] = bi;
    top1_p[n] = bp;
    int pos = atomicAdd(&counts[bi], 1);
    lists[bi * NTOK + pos] = n;
  }
}

// exclusive scan of counts + aux loss
__global__ void k_scan_aux(const int* __restrict__ counts, int* __restrict__ offs,
                           float* __restrict__ auxout) {
  if (blockIdx.x == 0 && threadIdx.x == 0) {
    int o = 0; float aux = 0.f;
    for (int e = 0; e < NEXP; ++e) {
      offs[e] = o; o += counts[e];
      float f = (float)counts[e] / (float)NTOK;
      aux += f * f;
    }
    offs[NEXP] = o;
    auxout[0] = aux * (float)NEXP;
  }
}

// ---------------------------------------------------------------------------
extern "C" void kernel_launch(void* const* d_in, const int* in_sizes, int n_in,
                              void* d_out, int out_size, void* d_ws, size_t ws_size,
                              hipStream_t stream) {
  (void)in_sizes; (void)n_in; (void)out_size; (void)ws_size;
  const float* x    = (const float*)d_in[0];
  const unsigned char* mask = (const unsigned char*)d_in[1];
  const float* Wq = (const float*)d_in[2];  const float* bq = (const float*)d_in[3];
  const float* Wk = (const float*)d_in[4];  const float* bk = (const float*)d_in[5];
  const float* Wv = (const float*)d_in[6];  const float* bv = (const float*)d_in[7];
  const float* Wo = (const float*)d_in[8];  const float* bo = (const float*)d_in[9];
  const float* ln1a = (const float*)d_in[10];
  const float* ln1b = (const float*)d_in[11];
  // d_in[12], d_in[13] = ln2 (computed-but-discarded in reference)
  const float* gW = (const float*)d_in[14]; const float* gb = (const float*)d_in[15];
  const float* W1 = (const float*)d_in[16]; const float* b1 = (const float*)d_in[17];
  const float* W2 = (const float*)d_in[18]; const float* b2 = (const float*)d_in[19];
  float* out = (float*)d_out;               // [N*D] x2, then [1] aux

  // ---- workspace carve-up (256B aligned)
  char* wsb = (char*)d_ws;
  size_t off = 0;
  auto take = [&](size_t bytes) -> char* {
    char* p = wsb + off;
    off += bytes; off = (off + 255) & ~(size_t)255;
    return p;
  };
  bf16_t* xnh  = (bf16_t*)take((size_t)NTOK * DMODEL * 2);
  bf16_t* Wqh  = (bf16_t*)take((size_t)DMODEL * DMODEL * 2);
  bf16_t* Wkh  = (bf16_t*)take((size_t)DMODEL * DMODEL * 2);
  bf16_t* Wvh  = (bf16_t*)take((size_t)DMODEL * DMODEL * 2);
  bf16_t* Woh  = (bf16_t*)take((size_t)DMODEL * DMODEL * 2);
  bf16_t* Qh   = (bf16_t*)take((size_t)NTOK * DMODEL * 2);
  bf16_t* Kh   = (bf16_t*)take((size_t)NTOK * DMODEL * 2);
  bf16_t* Vh   = (bf16_t*)take((size_t)NTOK * DMODEL * 2);
  bf16_t* attnh= (bf16_t*)take((size_t)NTOK * DMODEL * 2);
  float*  x1f  = (float*) take((size_t)NTOK * DMODEL * 4);
  bf16_t* x1h  = (bf16_t*)take((size_t)NTOK * DMODEL * 2);
  bf16_t* W1h  = (bf16_t*)take((size_t)NEXP * DMODEL * DFFN * 2);
  bf16_t* W2h  = (bf16_t*)take((size_t)NEXP * DFFN * DMODEL * 2);
  bf16_t* hbuf = (bf16_t*)take((size_t)(NTOK + 128) * DFFN * 2); // +128 row pad
  int*    tidx = (int*)   take((size_t)NTOK * 4);
  float*  tp   = (float*) take((size_t)NTOK * 4);
  int*    counts = (int*) take(64);
  int*    offs   = (int*) take(64);
  int*    lists  = (int*) take((size_t)NEXP * NTOK * 4);

  auto cvt = [&](const float* s, bf16_t* dst, size_t n) {
    int blocks = (int)((n + 1023) / 1024);
    k_cvt<<<blocks, 256, 0, stream>>>(s, dst, n);
  };

  // 0) zero routing counts (ws is poisoned, never re-zeroed by harness)
  k_zero_i32<<<1, 64, 0, stream>>>(counts, NEXP);

  // 1) weights -> bf16
  cvt(Wq, Wqh, (size_t)DMODEL * DMODEL);
  cvt(Wk, Wkh, (size_t)DMODEL * DMODEL);
  cvt(Wv, Wvh, (size_t)DMODEL * DMODEL);
  cvt(Wo, Woh, (size_t)DMODEL * DMODEL);
  cvt(W1, W1h, (size_t)NEXP * DMODEL * DFFN);
  cvt(W2, W2h, (size_t)NEXP * DFFN * DMODEL);

  // 2) LN1 -> bf16
  k_ln<<<NTOK, 256, 0, stream>>>(x, ln1a, ln1b, xnh);

  // 3) Q/K/V projections (WMMA, 128x128 tiles)
  dim3 gproj(NTOK / 128, DMODEL / 128, 1);
  k_gemm<0><<<gproj, 256, 0, stream>>>(xnh, DMODEL, Wqh, DMODEL, bq,
      NTOK, DMODEL, DMODEL, nullptr, nullptr, Qh,
      nullptr, nullptr, nullptr, nullptr, nullptr, nullptr, nullptr, nullptr, 0);
  k_gemm<0><<<gproj, 256, 0, stream>>>(xnh, DMODEL, Wkh, DMODEL, bk,
      NTOK, DMODEL, DMODEL, nullptr, nullptr, Kh,
      nullptr, nullptr, nullptr, nullptr, nullptr, nullptr, nullptr, nullptr, 0);
  k_gemm<0><<<gproj, 256, 0, stream>>>(xnh, DMODEL, Wvh, DMODEL, bv,
      NTOK, DMODEL, DMODEL, nullptr, nullptr, Vh,
      nullptr, nullptr, nullptr, nullptr, nullptr, nullptr, nullptr, nullptr, 0);

  // 4) fused flash attention (WMMA QK^T and PV)
  k_attn<<<dim3(SEQT / 64, (NTOK / SEQT) * NHEAD), 128, 0, stream>>>(
      Qh, Kh, Vh, mask, attnh);

  // 5) output projection + residual: x1 = x + attn @ Wo + bo  (f32 + bf16)
  k_gemm<0><<<gproj, 256, 0, stream>>>(attnh, DMODEL, Woh, DMODEL, bo,
      NTOK, DMODEL, DMODEL, x, x1f, x1h,
      nullptr, nullptr, nullptr, nullptr, nullptr, nullptr, nullptr, nullptr, 0);

  // 6) gate + routing
  k_gate<<<NTOK, 256, 0, stream>>>(x1f, gW, gb, tidx, tp, counts, lists);
  k_scan_aux<<<1, 32, 0, stream>>>(counts, offs, out + (size_t)NTOK * DMODEL);

  // 7) expert GEMM1: h = relu(gather(x1) @ W1[e] + b1[e])
  k_gemm<1><<<dim3(NTOK / 128, DFFN / 128, NEXP), 256, 0, stream>>>(
      nullptr, DMODEL, W1h, DFFN, b1, NTOK, DFFN, DMODEL,
      nullptr, nullptr, nullptr, counts, offs, lists, tp, hbuf, x1h,
      nullptr, nullptr, NTOK);

  // 8) expert GEMM2 + scatter: out[tok] = x1[tok] + p * (h @ W2[e] + b2[e])
  k_gemm<2><<<dim3(NTOK / 128, DMODEL / 128, NEXP), 256, 0, stream>>>(
      nullptr, DFFN, W2h, DMODEL, b2, NTOK, DMODEL, DFFN,
      nullptr, nullptr, nullptr, counts, offs, lists, tp, hbuf, nullptr,
      x1f, out, NTOK);
}
